// ProbTravelTime_gnn_25134148616283
// MI455X (gfx1250) — compile-verified, matrix operand order checked by reading, not run on hardware
//
#include <hip/hip_runtime.h>
#include <hip/hip_bf16.h>
#include <cstdint>

typedef __attribute__((ext_vector_type(16))) _Float16 v16h;
typedef __attribute__((ext_vector_type(4)))  _Float16 v4h;
typedef __attribute__((ext_vector_type(8)))  float    v8f;

#define TS   128          // tokens per block
#define SEQ  2048
#define LDX  136          // padded leading dim (halfs) for 128-wide activations
#define LDH  264          // padded leading dim for 256-wide H2

// ---------------------------------------------------------------------------
// Pack an f32 weight matrix W[K][N] (row-major, K = fan_in) into WMMA
// B-fragment order for v_wmma_f32_16x16x32_f16:
//   fragment(nt, kt) = 32 lanes x 16 halfs, stored contiguously so a lane's
//   fragment load is one aligned 32-byte read.
// Lane mapping (mirrors the 16-bit A-matrix K layout): N = lane&15,
// K = kt*32 + ((lane>=16)?8:0) + slot pattern.
// ---------------------------------------------------------------------------
__global__ __launch_bounds__(256) void pack_w_kernel(const float* __restrict__ W,
                                                     _Float16* __restrict__ out,
                                                     int K, int N) {
    int idx  = blockIdx.x * blockDim.x + threadIdx.x;
    int numK = K >> 5, numN = N >> 4;
    int total = (numK * numN) << 5;
    if (idx >= total) return;
    int lane = idx & 31;
    int frag = idx >> 5;
    int kt   = frag % numK;
    int nt   = frag / numK;
    int n     = (nt << 4) + (lane & 15);
    int kbase = (kt << 5) + ((lane >> 4) << 3);
    _Float16* o = out + (size_t)idx * 16;
#pragma unroll
    for (int v2 = 0; v2 < 8; ++v2) {
        int Kk = kbase + ((v2 >> 2) << 4) + ((v2 & 3) << 1);
        o[2 * v2]     = (_Float16)W[(size_t)Kk * N + n];
        o[2 * v2 + 1] = (_Float16)W[(size_t)(Kk + 1) * N + n];
    }
}

// Load a 16x32 f16 A fragment from LDS (row-major, leading dim LD halfs).
template <int LD>
__device__ inline v16h load_a_frag(const _Float16* __restrict__ smem, int mBase, int kBase) {
    int lane = threadIdx.x & 31;
    const _Float16* p = smem + (size_t)(mBase + (lane & 15)) * LD + kBase + ((lane >> 4) << 3);
    v16h a;
#pragma unroll
    for (int v2 = 0; v2 < 8; ++v2) {
        int Kk = ((v2 >> 2) << 4) + ((v2 & 3) << 1);
        a[2 * v2]     = p[Kk];
        a[2 * v2 + 1] = p[Kk + 1];
    }
    return a;
}

// Load a pre-packed 32x16 f16 B fragment (one aligned 32B read per lane).
__device__ inline v16h load_b_frag(const _Float16* __restrict__ packed, int numK, int nt, int kt) {
    int lane = threadIdx.x & 31;
    return ((const v16h*)packed)[(((size_t)nt * numK + kt) << 5) + lane];
}

__device__ inline v8f wmma_f16(v16h a, v16h b, v8f c) {
    return __builtin_amdgcn_wmma_f32_16x16x32_f16(false, a, false, b, (short)0, c, false, false);
}

// Branch-free SELU: scale*max(x,0) + scale*alpha*(expm1(min(x,0)))
__device__ inline float selu_f(float x) {
    return 1.0507009873554805f * fmaxf(x, 0.f)
         + 1.7580993408473766f * (__expf(fminf(x, 0.f)) - 1.f);
}

// ---------------------------------------------------------------------------
// Main fused kernel: gather + f2 MLP + f MLP + heads. One block = 128 tokens.
// ---------------------------------------------------------------------------
__global__ __launch_bounds__(256) void gnn_main_kernel(
    const float* __restrict__ rho, const float* __restrict__ cmat,
    const int* __restrict__ roads,
    const _Float16* __restrict__ pW2a, const float* __restrict__ f2_b1,
    const _Float16* __restrict__ pW2b, const float* __restrict__ f2_b2,
    const _Float16* __restrict__ pW1,  const float* __restrict__ f_b1,
    const float* __restrict__ f_W21, const float* __restrict__ f_b21,
    const float* __restrict__ f_W22, const float* __restrict__ f_b22,
    float* __restrict__ logm, float* __restrict__ logv)
{
    __shared__ _Float16 Xrho[TS * LDX];   // rho tile, f16
    __shared__ _Float16 Xc[TS * LDX];     // gathered c tile -> later c_t
    __shared__ _Float16 H2[TS * LDH];     // f2 hidden (256)
    __shared__ int sroads[TS];

    const int tid  = threadIdx.x;
    const int lane = tid & 31;
    const int wv   = tid >> 5;
    const int b    = blockIdx.x >> 4;          // S/TS = 16 tiles per batch row
    const int s0   = (blockIdx.x & 15) << 7;

    if (tid < TS) sroads[tid] = roads[(size_t)b * SEQ + s0 + tid];
    __syncthreads();

    // ---- gather rho and c rows into LDS as f16 (float4 loads, v4h stores) ----
    const float4* rho4 = (const float4*)(rho + (((size_t)b * SEQ + s0) << 7));
    const float4* c4   = (const float4*)cmat;
    for (int i = tid; i < TS * 32; i += 256) {
        int tok = i >> 5, col = (i & 31) << 2;
        float4 v = rho4[i];
        *(v4h*)&Xrho[(size_t)tok * LDX + col] =
            (v4h){(_Float16)v.x, (_Float16)v.y, (_Float16)v.z, (_Float16)v.w};
        int r = sroads[tok];
        float4 u = {0.f, 0.f, 0.f, 0.f};
        if (r != 0) u = c4[(size_t)(r - 1) * 32 + (i & 31)];
        *(v4h*)&Xc[(size_t)tok * LDX + col] =
            (v4h){(_Float16)u.x, (_Float16)u.y, (_Float16)u.z, (_Float16)u.w};
    }
    __syncthreads();

    const int mBase   = wv << 4;                      // wave owns one 16-token M tile
    const int rowBase = mBase + ((lane >> 4) << 3);   // D-tile row origin for this lane half

    // ---- f2 layer 1: H2 = selu(Xc @ W1 + b1), K=128, N=256 ----
    {
        v16h aC[4];
#pragma unroll
        for (int kt = 0; kt < 4; ++kt) aC[kt] = load_a_frag<LDX>(Xc, mBase, kt << 5);
        for (int nt = 0; nt < 16; ++nt) {
            v8f d = {};
#pragma unroll
            for (int kt = 0; kt < 4; ++kt)
                d = wmma_f16(aC[kt], load_b_frag(pW2a, 4, nt, kt), d);
            int n = (nt << 4) + (lane & 15);
            float bias = f2_b1[n];
#pragma unroll
            for (int r2 = 0; r2 < 8; ++r2)
                H2[(size_t)(rowBase + r2) * LDH + n] = (_Float16)selu_f(d[r2] + bias);
        }
    }
    __syncthreads();

    // ---- f2 layer 2: c_t = H2 @ W2 + b2, K=256, N=128 (overwrites Xc) ----
    {
        v16h aH[8];
#pragma unroll
        for (int kt = 0; kt < 8; ++kt) aH[kt] = load_a_frag<LDH>(H2, mBase, kt << 5);
        for (int nt = 0; nt < 8; ++nt) {
            v8f d = {};
#pragma unroll
            for (int kt = 0; kt < 8; ++kt)
                d = wmma_f16(aH[kt], load_b_frag(pW2b, 8, nt, kt), d);
            int n = (nt << 4) + (lane & 15);
            float bias = f2_b2[n];
#pragma unroll
            for (int r2 = 0; r2 < 8; ++r2)
                Xc[(size_t)(rowBase + r2) * LDX + n] = (_Float16)(d[r2] + bias);
        }
    }
    __syncthreads();

    // ---- f layer 1 fused with heads: h1 = relu([rho|c_t] @ fW1 + b1);
    //      logm = h1.W21 + b21, logv = h1.W22 + b22 (H1 never stored) ----
    float accM[8], accV[8];
#pragma unroll
    for (int r2 = 0; r2 < 8; ++r2) { accM[r2] = 0.f; accV[r2] = 0.f; }

    {
        v16h aX[8];
#pragma unroll
        for (int kt = 0; kt < 4; ++kt) {
            aX[kt]     = load_a_frag<LDX>(Xrho, mBase, kt << 5);
            aX[kt + 4] = load_a_frag<LDX>(Xc,   mBase, kt << 5);
        }
        for (int nt = 0; nt < 32; ++nt) {
            v8f d = {};
#pragma unroll
            for (int kt = 0; kt < 8; ++kt)
                d = wmma_f16(aX[kt], load_b_frag(pW1, 8, nt, kt), d);
            int n = (nt << 4) + (lane & 15);
            float bias = f_b1[n], w1 = f_W21[n], w2 = f_W22[n];
#pragma unroll
            for (int r2 = 0; r2 < 8; ++r2) {
                float h = fmaxf(d[r2] + bias, 0.f);
                accM[r2] += h * w1;
                accV[r2] += h * w2;
            }
        }
    }
    // reduce over the 16 lanes holding the N dimension of each D tile half
#pragma unroll
    for (int r2 = 0; r2 < 8; ++r2) {
#pragma unroll
        for (int off = 8; off > 0; off >>= 1) {
            accM[r2] += __shfl_xor(accM[r2], off, 16);
            accV[r2] += __shfl_xor(accV[r2], off, 16);
        }
    }
    if ((lane & 15) == 0) {
        size_t base = (size_t)b * SEQ + s0 + rowBase;
        float bm = f_b21[0], bv = f_b22[0];
#pragma unroll
        for (int r2 = 0; r2 < 8; ++r2) {
            logm[base + r2] = accM[r2] + bm;
            logv[base + r2] = accV[r2] + bv;
        }
    }
}

// ---------------------------------------------------------------------------
// Per-batch-row stable logsumexp over S=2048, then final outputs.
// out[0..63] = log_mu, out[64..127] = log_lam.
// ---------------------------------------------------------------------------
__global__ __launch_bounds__(256) void gnn_reduce_kernel(
    const float* __restrict__ logm, const float* __restrict__ logv,
    const float* __restrict__ w, const float* __restrict__ l,
    float* __restrict__ out)
{
    __shared__ float red[256];
    const int b = blockIdx.x, tid = threadIdx.x;
    float a1[8], a2[8];
#pragma unroll
    for (int i = 0; i < 8; ++i) {
        int s = tid + (i << 8);
        size_t idx = (size_t)b * SEQ + s;
        float lw = __logf(w[idx]);
        a1[i] = logm[idx] + lw;
        a2[i] = logv[idx] + 2.f * lw;
    }
    float m1 = a1[0], m2 = a2[0];
#pragma unroll
    for (int i = 1; i < 8; ++i) { m1 = fmaxf(m1, a1[i]); m2 = fmaxf(m2, a2[i]); }

    red[tid] = m1; __syncthreads();
    for (int st = 128; st > 0; st >>= 1) { if (tid < st) red[tid] = fmaxf(red[tid], red[tid + st]); __syncthreads(); }
    m1 = red[0]; __syncthreads();
    red[tid] = m2; __syncthreads();
    for (int st = 128; st > 0; st >>= 1) { if (tid < st) red[tid] = fmaxf(red[tid], red[tid + st]); __syncthreads(); }
    m2 = red[0]; __syncthreads();

    float s1 = 0.f, s2 = 0.f;
#pragma unroll
    for (int i = 0; i < 8; ++i) { s1 += __expf(a1[i] - m1); s2 += __expf(a2[i] - m2); }
    red[tid] = s1; __syncthreads();
    for (int st = 128; st > 0; st >>= 1) { if (tid < st) red[tid] += red[tid + st]; __syncthreads(); }
    s1 = red[0]; __syncthreads();
    red[tid] = s2; __syncthreads();
    for (int st = 128; st > 0; st >>= 1) { if (tid < st) red[tid] += red[tid + st]; __syncthreads(); }
    s2 = red[0];

    if (tid == 0) {
        float magg = m1 + __logf(s1);
        float vagg = m2 + __logf(s2);
        float logl = __logf(l[b]);
        out[b]      = logl - magg;
        out[64 + b] = logl - 3.f * magg - vagg;
    }
}

extern "C" void kernel_launch(void* const* d_in, const int* in_sizes, int n_in,
                              void* d_out, int out_size, void* d_ws, size_t ws_size,
                              hipStream_t stream) {
    const float* rho    = (const float*)d_in[0];
    const float* cmat   = (const float*)d_in[1];
    const float* w      = (const float*)d_in[2];
    const float* l      = (const float*)d_in[3];
    const int*   roads  = (const int*)d_in[4];
    const float* f2_W1  = (const float*)d_in[5];
    const float* f2_b1  = (const float*)d_in[6];
    const float* f2_W2  = (const float*)d_in[7];
    const float* f2_b2  = (const float*)d_in[8];
    const float* f_W1   = (const float*)d_in[9];
    const float* f_b1   = (const float*)d_in[10];
    const float* f_W21  = (const float*)d_in[11];
    const float* f_b21  = (const float*)d_in[12];
    const float* f_W22  = (const float*)d_in[13];
    const float* f_b22  = (const float*)d_in[14];

    char* ws = (char*)d_ws;
    _Float16* pW2a = (_Float16*)(ws);             // 128x256 f16 packed  (64 KB)
    _Float16* pW2b = (_Float16*)(ws + 65536);     // 256x128 f16 packed  (64 KB)
    _Float16* pW1  = (_Float16*)(ws + 131072);    // 256x512 f16 packed  (256 KB)
    float*    logm = (float*)(ws + 393216);       // 64*2048 f32 (512 KB)
    float*    logv = (float*)(ws + 917504);       // 64*2048 f32 (512 KB)

    pack_w_kernel<<<8,  256, 0, stream>>>(f2_W1, pW2a, 128, 256);
    pack_w_kernel<<<8,  256, 0, stream>>>(f2_W2, pW2b, 256, 128);
    pack_w_kernel<<<32, 256, 0, stream>>>(f_W1,  pW1,  256, 512);

    gnn_main_kernel<<<1024, 256, 0, stream>>>(rho, cmat, roads,
                                              pW2a, f2_b1, pW2b, f2_b2,
                                              pW1, f_b1, f_W21, f_b21, f_W22, f_b22,
                                              logm, logv);

    gnn_reduce_kernel<<<64, 256, 0, stream>>>(logm, logv, w, l, (float*)d_out);
}
